// ActLayer_49641232007349
// MI455X (gfx1250) — matrix operand order repaired and checked by reading, not compiled, not run on hardware
//
#include <hip/hip_runtime.h>
#include <hip/hip_bf16.h>

typedef __attribute__((ext_vector_type(16))) _Float16 v16h;
typedef __attribute__((ext_vector_type(8)))  _Float16 v8h;
typedef __attribute__((ext_vector_type(8)))  float    v8f;

#define B_DIM   8192
#define I_DIM   512
#define F_DIM   64
#define O_DIM   512
#define K_DIM   (I_DIM * F_DIM)          // 32768
#define BT      64                        // batch-rows per workgroup
#define WC_BYTES ((size_t)O_DIM * K_DIM * sizeof(_Float16))  // 32 MB

// ---------------------------------------------------------------------------
// Kernel 1: build Wc[o][k] = lamb[i,o] * beta[f,o] * invstd[f]  (f16, o-major)
// and epi[o] = bias[o] - (sum_f mean_f*invstd_f*beta[f,o]) * (sum_i lamb[i,o])
// ---------------------------------------------------------------------------
__global__ __launch_bounds__(256)
void ActLayer_prep_kernel(const float* __restrict__ freqs,
                          const float* __restrict__ phases,
                          const float* __restrict__ beta,
                          const float* __restrict__ lamb,
                          const float* __restrict__ bias,
                          _Float16* __restrict__ Wc,
                          float* __restrict__ epi)
{
    __shared__ float sBN[F_DIM];
    __shared__ float sM[F_DIM];
    __shared__ float sL[256];

    const int o = blockIdx.x;
    const int t = threadIdx.x;

    if (t < F_DIM) {
        float w  = freqs[t];
        float ph = phases[t];
        float mean = __expf(-0.5f * w * w) * __sinf(ph);
        float var  = 0.5f - 0.5f * __expf(-2.0f * w * w) * __cosf(2.0f * ph)
                     - mean * mean;
        float inv  = rsqrtf(1e-3f + var);
        float bN   = beta[t * O_DIM + o] * inv;
        sBN[t] = bN;
        sM[t]  = mean * bN;
    }
    sL[t] = lamb[(size_t)t * O_DIM + o] + lamb[(size_t)(t + 256) * O_DIM + o];
    __syncthreads();

    if (t == 0) {
        float ms = 0.0f, ls = 0.0f;
        for (int f = 0; f < F_DIM; ++f) ms += sM[f];
        for (int j = 0; j < 256; ++j)   ls += sL[j];
        epi[o] = bias[o] - ms * ls;
    }

    _Float16* wrow = Wc + (size_t)o * K_DIM;
    for (int kb = t; kb < K_DIM; kb += 256) {
        int i = kb >> 6;
        int f = kb & 63;
        wrow[kb] = (_Float16)(lamb[(size_t)i * O_DIM + o] * sBN[f]);
    }
}

// ---------------------------------------------------------------------------
// Kernel 2: fused feature generation + GEMM.
// Grid: B/64 = 128 workgroups, 256 threads (8 waves).
// Each WG: rows [b0, b0+64), all 512 output columns.
// Per wave: 4 row-tiles x 4 col-tiles of 16x16 f32 accumulators.
// ---------------------------------------------------------------------------
__global__ __launch_bounds__(256)
void ActLayer_49641232007349_kernel(const float* __restrict__ x,
                                    const float* __restrict__ freqs,
                                    const float* __restrict__ phases,
                                    const _Float16* __restrict__ Wc,
                                    const float* __restrict__ epi,
                                    float* __restrict__ out)
{
    // Ping-pong feature tile: 64 rows x 64 f (padded to 72 halves => 36-dword
    // row stride, conflict-free b128 LDS reads). 2*64*72*2 = 18 KB.
    __shared__ __align__(16) _Float16 ash[2][BT][72];

    const int t  = threadIdx.x;
    const int b0 = blockIdx.x * BT;

    // --- generation mapping: 4 threads per row, 16 freqs per thread ---
    const int grow = t >> 2;            // 0..63
    const int gf   = (t & 3) << 4;      // 0,16,32,48
    float w[16], p[16];
#pragma unroll
    for (int j = 0; j < 16; ++j) {
        w[j] = freqs[gf + j];
        p[j] = phases[gf + j];
    }
    const float* xrow = x + (size_t)(b0 + grow) * I_DIM;

    // --- consumption mapping ---
    const int lane = t & 31;
    const int wv   = t >> 5;            // wave 0..7 -> col-tiles wv*4 .. wv*4+3
    const int hi   = lane >> 4;         // half-wave
    const int lr   = lane & 15;

    v8f acc[4][4];
    const v8f vzero = {0, 0, 0, 0, 0, 0, 0, 0};
#pragma unroll
    for (int rt = 0; rt < 4; ++rt)
#pragma unroll
        for (int c = 0; c < 4; ++c) acc[rt][c] = vzero;

    // Prologue: generate features for i = 0 into buffer 0.
    {
        float xv = xrow[0];
        v8h t0, t1;
#pragma unroll
        for (int j = 0; j < 8; ++j) {
            t0[j] = (_Float16)__sinf(w[j]     * xv + p[j]);
            t1[j] = (_Float16)__sinf(w[j + 8] * xv + p[j + 8]);
        }
        *(v8h*)&ash[0][grow][gf]     = t0;
        *(v8h*)&ash[0][grow][gf + 8] = t1;
    }
    __syncthreads();

    for (int i = 0; i < I_DIM; ++i) {
        const int cur = i & 1;

        // Generate next i's features into the other buffer (overlaps WMMA).
        if (i + 1 < I_DIM) {
            float xv = xrow[i + 1];
            v8h t0, t1;
#pragma unroll
            for (int j = 0; j < 8; ++j) {
                t0[j] = (_Float16)__sinf(w[j]     * xv + p[j]);
                t1[j] = (_Float16)__sinf(w[j + 8] * xv + p[j + 8]);
            }
            *(v8h*)&ash[cur ^ 1][grow][gf]     = t0;
            *(v8h*)&ash[cur ^ 1][grow][gf + 8] = t1;
        }

        // Consume current buffer: two K=32 steps (f 0..31, 32..63).
#pragma unroll
        for (int s = 0; s < 2; ++s) {
            v16h afrag[4];
#pragma unroll
            for (int rt = 0; rt < 4; ++rt) {
                // 16-bit A 16x32 layout: lo lanes K {0..7,16..23}, hi lanes
                // K {8..15,24..31} relative to this step's base.
                const _Float16* ap = &ash[cur][rt * 16 + lr][s * 32 + hi * 8];
                v8h a0 = *(const v8h*)ap;
                v8h a1 = *(const v8h*)(ap + 16);
                afrag[rt] = __builtin_shufflevector(
                    a0, a1, 0, 1, 2, 3, 4, 5, 6, 7, 8, 9, 10, 11, 12, 13, 14, 15);
            }
#pragma unroll
            for (int c = 0; c < 4; ++c) {
                // B 32x16 layout: lo lanes K {0..7,16..23}, hi lanes K {8..15,24..31}, col=lr.
                const _Float16* bp = Wc
                    + (size_t)((wv * 4 + c) * 16 + lr) * K_DIM
                    + (size_t)i * F_DIM + s * 32 + hi * 8;
                v8h q0 = *(const v8h*)bp;
                v8h q1 = *(const v8h*)(bp + 16);
                v16h bfrag = __builtin_shufflevector(
                    q0, q1, 0, 1, 2, 3, 4, 5, 6, 7, 8, 9, 10, 11, 12, 13, 14, 15);
#pragma unroll
                for (int rt = 0; rt < 4; ++rt)
                    acc[rt][c] = __builtin_amdgcn_wmma_f32_16x16x32_f16(
                        false, afrag[rt], false, bfrag,
                        (short)0, acc[rt][c], false, false);
            }
        }
        __syncthreads();
    }

    // Epilogue: C/D layout -> VGPR v holds M = v + 8*hi, N = lr within tile.
#pragma unroll
    for (int c = 0; c < 4; ++c) {
        const int col = (wv * 4 + c) * 16 + lr;
        const float e = epi[col];
#pragma unroll
        for (int rt = 0; rt < 4; ++rt) {
#pragma unroll
            for (int v = 0; v < 8; ++v) {
                const int r = b0 + rt * 16 + hi * 8 + v;
                out[(size_t)r * O_DIM + col] = acc[rt][c][v] + e;
            }
        }
    }
}

extern "C" void kernel_launch(void* const* d_in, const int* in_sizes, int n_in,
                              void* d_out, int out_size, void* d_ws, size_t ws_size,
                              hipStream_t stream) {
    const float* x      = (const float*)d_in[0];
    const float* freqs  = (const float*)d_in[1];
    const float* phases = (const float*)d_in[2];
    const float* beta   = (const float*)d_in[3];
    const float* lamb   = (const float*)d_in[4];
    const float* bias   = (const float*)d_in[5];
    float*       out    = (float*)d_out;

    _Float16* Wc  = (_Float16*)d_ws;                         // 32 MB, L2-resident
    float*    epi = (float*)((char*)d_ws + WC_BYTES);        // 2 KB

    ActLayer_prep_kernel<<<O_DIM, 256, 0, stream>>>(freqs, phases, beta, lamb,
                                                    bias, Wc, epi);
    ActLayer_49641232007349_kernel<<<B_DIM / BT, 256, 0, stream>>>(
        x, freqs, phases, Wc, epi, out);
}
